// ImgNetHY_16853451669864
// MI455X (gfx1250) — matrix operand order repaired and checked by reading, not compiled, not run on hardware
//
#include <hip/hip_runtime.h>

typedef __attribute__((ext_vector_type(16))) __bf16          v16bf;
typedef __attribute__((ext_vector_type(8)))  float           v8f;
typedef __attribute__((ext_vector_type(8)))  unsigned short  us8;

#define N_NODES 8192
#define FEATD   4096
#define BHID    4096
#define CODELEN 64
#define KUNION  576   // 512 spline + 64 base

// ---------------- helpers ----------------
__device__ __forceinline__ unsigned short f2bf(float f) {
  union { float f; unsigned int u; } v; v.f = f;
  unsigned int u = v.u;
  unsigned int rb = ((u >> 16) & 1u) + 0x7FFFu;   // round-to-nearest-even
  return (unsigned short)((u + rb) >> 16);
}

__global__ void cvt_f32_bf16(const float* __restrict__ src,
                             unsigned short* __restrict__ dst, size_t n) {
  size_t i = blockIdx.x * (size_t)blockDim.x + threadIdx.x;
  size_t stride = (size_t)gridDim.x * blockDim.x;
  for (; i < n; i += stride) dst[i] = f2bf(src[i]);
}

// WcatT[k][n] = k<512 ? Ws[n][k] : Wb[n][k-512]   (bf16), k in [0,576), n in [0,8192)
__global__ void build_wct(const float* __restrict__ Ws, const float* __restrict__ Wb,
                          unsigned short* __restrict__ WcT) {
  size_t i = blockIdx.x * (size_t)blockDim.x + threadIdx.x;
  size_t total = (size_t)KUNION * N_NODES;
  size_t stride = (size_t)gridDim.x * blockDim.x;
  for (; i < total; i += stride) {
    int k = (int)(i / N_NODES);
    int n = (int)(i % N_NODES);
    float v = (k < 512) ? Ws[(size_t)n * 512 + k] : Wb[(size_t)n * 64 + (k - 512)];
    WcT[i] = f2bf(v);
  }
}

// ---------------- tiled bf16 WMMA GEMM, ping-pong LDS pipeline ----------------
// C[M,N] = op(A[M,K] @ B[K,N] (+ bias[N]))   A,B bf16 row-major, acc f32
// Requires: M % 128 == 0, K % 32 == 0, N % 16 == 0 (N may be < 128)
#define BM 128
#define BN 128
#define BK 32
#define LBS 40                 // k-stride (u16) of transposed B tile
#define LBSZ (BN * LBS + 64)   // + room for per-group skew

// B^T tile address: column c, k index k.  8-u16 skew per 16-column group keeps
// the 16-col-apart store lanes on distinct banks (stride 324 dw = 4 mod 64),
// while fragment reads (c within one group) stay 16B-aligned & conflict-free.
__device__ __forceinline__ int bidx(int c, int k) {
  return c * LBS + ((c >> 4) << 3) + k;
}

template<bool RELU, bool BIAS, bool OUTBF16>
__global__ __launch_bounds__(256, 2)
void gemm_bf16_wmma(const unsigned short* __restrict__ A,
                    const unsigned short* __restrict__ B,
                    const float* __restrict__ bias,
                    void* __restrict__ Cout,
                    int M, int N, int K) {
  __shared__ __align__(16) unsigned short lA[2][BM * BK];   // [m][k]
  __shared__ __align__(16) unsigned short lB[2][LBSZ];      // [n][k] transposed+skewed

  const int tid  = threadIdx.x;
  const int lane = tid & 31;
  const int wid  = tid >> 5;        // 8 waves
  const int wm   = wid >> 2;        // 0..1 -> 64-row strip
  const int wn   = wid & 3;         // 0..3 -> 32-col strip
  const int bm0  = blockIdx.y * BM;
  const int bn0  = blockIdx.x * BN;
  const int l16  = lane & 15;
  const int lhi  = lane >> 4;       // 0 or 1

  const int arow  = tid >> 1;           // A stage: 2 threads per row
  const int apart = (tid & 1) << 4;     // 16 bf16 each
  const int bkrow = tid >> 3;           // B stage: 8 threads per k-row
  const int bseg  = (tid & 7) << 4;     // 16 cols each
  const int gcol0 = bn0 + bseg;
  const bool bIn  = (gcol0 < N);        // N % 16 == 0 -> segment fully in or fully out

  const unsigned short* gaBase = A + (size_t)(bm0 + arow) * K + apart;

  auto loadTiles = [&](int k0, us8& a0, us8& a1, us8& b0, us8& b1) {
    const unsigned short* ga = gaBase + k0;
    a0 = *(const us8*)(ga);
    a1 = *(const us8*)(ga + 8);
    if (bIn) {
      const unsigned short* gb = B + (size_t)(k0 + bkrow) * N + gcol0;
      b0 = *(const us8*)(gb);
      b1 = *(const us8*)(gb + 8);
    } else {
      #pragma unroll
      for (int e = 0; e < 8; ++e) { b0[e] = 0; b1[e] = 0; }
    }
  };
  auto storeTiles = [&](int p, us8 a0, us8 a1, us8 b0, us8 b1) {
    *(us8*)&lA[p][arow * BK + apart]     = a0;
    *(us8*)&lA[p][arow * BK + apart + 8] = a1;
    #pragma unroll
    for (int e = 0; e < 8; ++e) {
      lB[p][bidx(bseg + e,     bkrow)] = b0[e];
      lB[p][bidx(bseg + 8 + e, bkrow)] = b1[e];
    }
  };

  v8f acc[4][2];
  #pragma unroll
  for (int i = 0; i < 4; ++i)
    #pragma unroll
    for (int j = 0; j < 2; ++j)
      #pragma unroll
      for (int e = 0; e < 8; ++e) acc[i][j][e] = 0.0f;

  us8 a0, a1, b0, b1;
  loadTiles(0, a0, a1, b0, b1);
  storeTiles(0, a0, a1, b0, b1);
  __syncthreads();

  int p = 0;
  for (int k0 = 0; k0 < K; k0 += BK, p ^= 1) {
    const bool more = (k0 + BK) < K;
    if (more) loadTiles(k0 + BK, a0, a1, b0, b1);   // HBM overlaps WMMA below

    // ---- fragments (ISA 16-bit A layout: lanes 0-15 K0-7/16-23, 16-31 K8-15/24-31)
    v16bf afrag[4], bfrag[2];
    #pragma unroll
    for (int ni = 0; ni < 2; ++ni) {
      const int c = wn * 32 + ni * 16 + l16;
      const int base = bidx(c, lhi * 8);
      union { us8 s[2]; v16bf v; } u;
      u.s[0] = *(const us8*)&lB[p][base];
      u.s[1] = *(const us8*)&lB[p][base + 16];
      bfrag[ni] = u.v;
    }
    #pragma unroll
    for (int mi = 0; mi < 4; ++mi) {
      const int r = wm * 64 + mi * 16 + l16;
      const int base = r * BK + lhi * 8;
      union { us8 s[2]; v16bf v; } u;
      u.s[0] = *(const us8*)&lA[p][base];
      u.s[1] = *(const us8*)&lA[p][base + 16];
      afrag[mi] = u.v;
    }

    #pragma unroll
    for (int mi = 0; mi < 4; ++mi)
      #pragma unroll
      for (int ni = 0; ni < 2; ++ni)
        acc[mi][ni] = __builtin_amdgcn_wmma_f32_16x16x32_bf16(
            false, afrag[mi], false, bfrag[ni], (short)0, acc[mi][ni], false, false);

    // commit next tile into the other buffer (last read k0-BK; barrier below covers it)
    if (more) storeTiles(p ^ 1, a0, a1, b0, b1);
    __syncthreads();
  }

  // ---- epilogue: C/D layout VGPR j -> M = base + j + 8*lhi, N = base + l16
  #pragma unroll
  for (int mi = 0; mi < 4; ++mi) {
    const int row0 = bm0 + wm * 64 + mi * 16 + lhi * 8;
    #pragma unroll
    for (int ni = 0; ni < 2; ++ni) {
      const int col = bn0 + wn * 32 + ni * 16 + l16;
      if (col < N) {
        const float bi = BIAS ? bias[col] : 0.0f;
        #pragma unroll
        for (int j = 0; j < 8; ++j) {
          float v = acc[mi][ni][j] + bi;
          if (RELU) v = v > 0.0f ? v : 0.0f;
          const size_t off = (size_t)(row0 + j) * N + col;
          if (OUTBF16) ((unsigned short*)Cout)[off] = f2bf(v);
          else         ((float*)Cout)[off] = v;
        }
      }
    }
  }
}

// ---------------- KAN: tanh -> code, LayerNorm -> RBF basis + silu -> U ----------------
__global__ __launch_bounds__(256)
void kan_rbf(const float* __restrict__ feat,
             const float* __restrict__ ln_g, const float* __restrict__ ln_b,
             float* __restrict__ code_out, unsigned short* __restrict__ U) {
  const int lane = threadIdx.x & 31;
  const int row  = blockIdx.x * 8 + (threadIdx.x >> 5);   // one wave32 per row of 64
  const float invden = 7.0f / 4.0f;                        // 1/DENOM, DENOM = 4/7

  float c[2];
  #pragma unroll
  for (int q = 0; q < 2; ++q) {
    float f = feat[(size_t)row * CODELEN + lane * 2 + q];
    c[q] = tanhf(10.0f * f);
    code_out[(size_t)row * CODELEN + lane * 2 + q] = c[q];
  }
  float s = c[0] + c[1], ss = c[0] * c[0] + c[1] * c[1];
  #pragma unroll
  for (int off = 16; off > 0; off >>= 1) {
    s  += __shfl_xor(s,  off, 32);
    ss += __shfl_xor(ss, off, 32);
  }
  const float mean = s * (1.0f / 64.0f);
  const float var  = ss * (1.0f / 64.0f) - mean * mean;
  const float rstd = rsqrtf(var + 1e-5f);

  #pragma unroll
  for (int q = 0; q < 2; ++q) {
    const int idx = lane * 2 + q;
    const float z = (c[q] - mean) * rstd * ln_g[idx] + ln_b[idx];
    us8 bvec;
    #pragma unroll
    for (int g = 0; g < 8; ++g) {
      const float gr = -2.0f + (float)g * (4.0f / 7.0f);
      const float t = (z - gr) * invden;
      bvec[g] = f2bf(__expf(-t * t));
    }
    *(us8*)&U[(size_t)row * KUNION + idx * 8] = bvec;            // spline basis block
    const float sig = 1.0f / (1.0f + __expf(-c[q]));
    U[(size_t)row * KUNION + 512 + idx] = f2bf(c[q] * sig);      // silu(base) block
  }
}

// ---------------- orchestration ----------------
extern "C" void kernel_launch(void* const* d_in, const int* in_sizes, int n_in,
                              void* d_out, int out_size, void* d_ws, size_t ws_size,
                              hipStream_t stream) {
  (void)in_sizes; (void)n_in; (void)out_size; (void)ws_size;
  const float* x   = (const float*)d_in[0];
  const float* G   = (const float*)d_in[1];
  const float* W1  = (const float*)d_in[2];
  const float* b1  = (const float*)d_in[3];
  const float* W2  = (const float*)d_in[4];
  const float* b2  = (const float*)d_in[5];
  const float* lng = (const float*)d_in[6];
  const float* lnb = (const float*)d_in[7];
  const float* Ws  = (const float*)d_in[8];
  const float* Wb  = (const float*)d_in[9];
  const float* bb  = (const float*)d_in[10];

  char* p = (char*)d_ws;
  auto take = [&](size_t bytes) { char* r = p; p += (bytes + 255) & ~(size_t)255; return r; };
  unsigned short* xb  = (unsigned short*)take((size_t)N_NODES * FEATD * 2);
  unsigned short* Gb  = (unsigned short*)take((size_t)N_NODES * N_NODES * 2);
  unsigned short* W1b = (unsigned short*)take((size_t)FEATD * BHID * 2);
  unsigned short* t1b = (unsigned short*)take((size_t)N_NODES * BHID * 2);
  unsigned short* W2b = (unsigned short*)take((size_t)BHID * CODELEN * 2);
  unsigned short* t2b = (unsigned short*)take((size_t)N_NODES * CODELEN * 2);
  float*          ft  = (float*)take((size_t)N_NODES * CODELEN * 4);
  unsigned short* Ub  = (unsigned short*)take((size_t)N_NODES * KUNION * 2);
  unsigned short* WcT = (unsigned short*)take((size_t)KUNION * N_NODES * 2);
  unsigned short* hb  = xb;   // x is dead after GEMM1 -> reuse for h

  float* code = (float*)d_out;
  float* out  = (float*)d_out + (size_t)N_NODES * CODELEN;

  dim3 blk(256);
  cvt_f32_bf16<<<4096, blk, 0, stream>>>(x,  xb,  (size_t)N_NODES * FEATD);
  cvt_f32_bf16<<<8192, blk, 0, stream>>>(G,  Gb,  (size_t)N_NODES * N_NODES);
  cvt_f32_bf16<<<2048, blk, 0, stream>>>(W1, W1b, (size_t)FEATD * BHID);
  cvt_f32_bf16<<<256,  blk, 0, stream>>>(W2, W2b, (size_t)BHID * CODELEN);
  build_wct<<<4096, blk, 0, stream>>>(Ws, Wb, WcT);

  // t1 = x@W1 + b1                                  [8192,4096] bf16
  gemm_bf16_wmma<false, true, true><<<dim3(BHID / BN, N_NODES / BM), blk, 0, stream>>>(
      xb, W1b, b1, t1b, N_NODES, BHID, FEATD);
  // h = relu(G@t1)                                  [8192,4096] bf16 (overwrites xb)
  gemm_bf16_wmma<true, false, true><<<dim3(BHID / BN, N_NODES / BM), blk, 0, stream>>>(
      Gb, t1b, nullptr, hb, N_NODES, BHID, N_NODES);
  // t2 = h@W2 + b2                                  [8192,64] bf16
  gemm_bf16_wmma<false, true, true><<<dim3(1, N_NODES / BM), blk, 0, stream>>>(
      hb, W2b, b2, t2b, N_NODES, CODELEN, BHID);
  // feat = G@t2                                     [8192,64] f32
  gemm_bf16_wmma<false, false, false><<<dim3(1, N_NODES / BM), blk, 0, stream>>>(
      Gb, t2b, nullptr, ft, N_NODES, CODELEN, N_NODES);
  // code (output 0) + fused KAN operand U = [basis | silu(code)]
  kan_rbf<<<N_NODES / 8, blk, 0, stream>>>(ft, lng, lnb, code, Ub);
  // out = relu(U @ WcatT + bb)                      [8192,8192] f32 (output 1)
  gemm_bf16_wmma<true, true, false><<<dim3(N_NODES / BN, N_NODES / BM), blk, 0, stream>>>(
      Ub, WcT, bb, out, N_NODES, N_NODES, KUNION);
}
